// WaveletAttention_1211180778054
// MI455X (gfx1250) — compile-verified, matrix operand order checked by reading, not run on hardware
//
#include <hip/hip_runtime.h>

#define BDIM 4
#define LDIM 2048
#define DDIM 512
#define HDIM 8
#define DH   64
// 1/sqrt(64) * log2(e), folded into Q so softmax runs in base-2 domain
#define QSCALE 0.18033688011112042f

typedef __attribute__((ext_vector_type(16))) _Float16 v16h;
typedef __attribute__((ext_vector_type(8)))  _Float16 h8;
typedef __attribute__((ext_vector_type(8)))  float    v8f;
typedef __attribute__((ext_vector_type(4)))  unsigned int u32x4;
typedef __attribute__((ext_vector_type(8)))  int i32x8;
typedef __attribute__((ext_vector_type(4)))  int i32x4;

union AFrag { v16h v; h8 h[2]; };

// ---------------------------------------------------------------------------
// TDM 2D tile load: build a D# (groups 0/1 per CDNA5 ISA) and
// issue TENSOR_LOAD_TO_LDS (6-arg builtin form). data_size=2B (f16), type=2.
// ---------------------------------------------------------------------------
__device__ __forceinline__ void tdm_load_2d(unsigned lds_addr, const void* gaddr,
                                            unsigned tensor_d0, unsigned tensor_d1,
                                            unsigned tile_d0, unsigned tile_d1,
                                            unsigned stride0) {
    unsigned long long ga = (unsigned long long)gaddr;
    u32x4 g0;
    g0[0] = 1u;                                   // count=1, user descriptor
    g0[1] = lds_addr;                             // bits 63:32  lds_addr
    g0[2] = (unsigned)(ga & 0xFFFFFFFFu);         // bits 95:64  global_addr[31:0]
    g0[3] = (unsigned)((ga >> 32) & 0x01FFFFFFu)  // bits 120:96 global_addr[56:32]
          | 0x80000000u;                          // bits 127:126 type=2
    i32x8 g1;
    g1[0] = (int)(1u << 16);                      // data_size=1 -> 2 bytes
    g1[1] = (int)((tensor_d0 & 0xFFFFu) << 16);   // tensor_dim0[15:0] @ bits 63:48
    g1[2] = (int)(((tensor_d0 >> 16) & 0xFFFFu)   // tensor_dim0[31:16]
          | ((tensor_d1 & 0xFFFFu) << 16));       // tensor_dim1[15:0]
    g1[3] = (int)(((tensor_d1 >> 16) & 0xFFFFu)   // tensor_dim1[31:16]
          | ((tile_d0 & 0xFFFFu) << 16));         // tile_dim0
    g1[4] = (int)(tile_d1 & 0xFFFFu);             // tile_dim1 (tile_dim2 = 0)
    g1[5] = (int)stride0;                         // tensor_dim0_stride[31:0]
    g1[6] = 0;
    g1[7] = 0;
    i32x4 z4 = {0, 0, 0, 0};
    i32x8 z8 = {0, 0, 0, 0, 0, 0, 0, 0};
    __builtin_amdgcn_tensor_load_to_lds(g0, g1, z4, z4, z8, 0);
}

// ---------------------------------------------------------------------------
// Kernel 1: 3-level MODWT cascade (db4, dilations 1/2/4) per (b, d) row.
// Dense [L,L] filter matrices are circulant with 8 taps (read from row 0).
// Q scaled by 0.125*log2(e); Q,K -> [B,H,L,64] f16, V -> [B,H,64,L] f16 (T).
// ---------------------------------------------------------------------------
__global__ __launch_bounds__(256)
void wavelet_qkv_kernel(const float* __restrict__ x,
                        const float* __restrict__ vfilt,
                        const float* __restrict__ wfilt,
                        _Float16* __restrict__ Q,
                        _Float16* __restrict__ K,
                        _Float16* __restrict__ VT) {
    __shared__ float buf0[LDIM];
    __shared__ float buf1[LDIM];
    const int tid = threadIdx.x;
    const int b = blockIdx.x >> 9;
    const int d = blockIdx.x & (DDIM - 1);
    const int h = d >> 6;
    const int c = d & (DH - 1);
    const long bh = (long)b * HDIM + h;

#pragma unroll
    for (int i = 0; i < 8; ++i) {
        int m = tid + i * 256;
        buf0[m] = x[((long)b * LDIM + m) * DDIM + d];
    }
    float* cur = buf0;
    float* nxt = buf1;
    __syncthreads();

    for (int j = 0; j < 3; ++j) {
        const int s = 1 << j;
        float wlo[8], whi[8];
#pragma unroll
        for (int k = 0; k < 8; ++k) {
            int idx = (LDIM - s * k) & (LDIM - 1);
            wlo[k] = vfilt[(long)j * LDIM * LDIM + idx];
            whi[k] = wfilt[(long)j * LDIM * LDIM + idx];
        }
#pragma unroll
        for (int i = 0; i < 8; ++i) {
            int m = tid + i * 256;
            float acc = 0.f;
#pragma unroll
            for (int k = 0; k < 8; ++k)
                acc += wlo[k] * cur[(m + LDIM - s * k) & (LDIM - 1)];
            nxt[m] = acc;
        }
        __syncthreads();
        { float* t = cur; cur = nxt; nxt = t; }
#pragma unroll
        for (int i = 0; i < 8; ++i) {
            int m = tid + i * 256;
            float acc = 0.f;
#pragma unroll
            for (int k = 0; k < 8; ++k)
                acc += whi[k] * cur[(m + LDIM - s * k) & (LDIM - 1)];
            if (j == 0)      Q[(bh * LDIM + m) * DH + c] = (_Float16)(acc * QSCALE);
            else if (j == 1) K[(bh * LDIM + m) * DH + c] = (_Float16)acc;
            else             VT[(bh * DH + c) * LDIM + m] = (_Float16)acc;
        }
        __syncthreads();
    }
}

// ---------------------------------------------------------------------------
// Kernel 2: flash attention with transposed score tiles.
//   S^T = K_tile x Q^T  -> C layout: regs = keys, lanes = queries
//   softmax: in-lane over 16 regs + one shfl_xor(16) across lane halves
//   out^T += V^T x P    -> P B-fragment is exactly the in-lane exp2 values
// K/V tiles double-buffered in LDS, fed asynchronously by the Tensor Data
// Mover: wave 0 issues tile kt+1, then s_wait_tensorcnt(2) guarantees only
// the older pair (tile kt) has landed -> DMA overlaps the 8 WMMAs + softmax.
// ---------------------------------------------------------------------------
__global__ __launch_bounds__(256)
void flash_attn_kernel(const _Float16* __restrict__ Q,
                       const _Float16* __restrict__ K,
                       const _Float16* __restrict__ VT,
                       float* __restrict__ out) {
    __shared__ __align__(16) _Float16 Kt[2][32 * DH]; // [buf][key 0..31][dh 0..63]
    __shared__ __align__(16) _Float16 Vt[2][DH * 32]; // [buf][dh 0..63][key 0..31]

    const int tid  = threadIdx.x;
    const int wave = tid >> 5;
    const int lane = tid & 31;
    const int ln   = lane & 15;
    const int hiL  = lane >> 4;
    const int kb0  = hiL * 8;                  // fragment K-chunk base per lane half

    const int bh = blockIdx.x >> 4;
    const int qb = blockIdx.x & 15;
    const int qrow0 = qb * 128 + wave * 16;

    const long qg = ((long)bh * LDIM + qrow0) * DH;
    const long kg = (long)bh * LDIM * DH;
    const long vg = (long)bh * DH * LDIM;

    // Q fragments (B operand of K x Q^T): lane = q row, chunks kb0 / kb0+16
    AFrag qf[2];
#pragma unroll
    for (int f = 0; f < 2; ++f) {
        const _Float16* p = Q + qg + (long)ln * DH + f * 32 + kb0;
        qf[f].h[0] = *(const h8*)(p);
        qf[f].h[1] = *(const h8*)(p + 16);
    }

    v8f acc[4];                                // out^T, 4 dh tiles of 16
#pragma unroll
    for (int n = 0; n < 4; ++n)
#pragma unroll
        for (int r = 0; r < 8; ++r) acc[n][r] = 0.f;
    float m = -1e30f, l = 0.f;                 // per-lane: stats of query `ln`

    const int NT = LDIM / 32;                  // 64 key tiles
    if (wave == 0) {                           // prologue: tile 0 -> buffer 0
        tdm_load_2d((unsigned)(unsigned long long)(void*)&Kt[0][0],
                    K + kg, DH, LDIM, DH, 32, DH);
        tdm_load_2d((unsigned)(unsigned long long)(void*)&Vt[0][0],
                    VT + vg, LDIM, DH, 32, DH, LDIM);
    }

    for (int kt = 0; kt < NT; ++kt) {
        const int cb = kt & 1;
        if (wave == 0) {
            // prefetch tile kt+1 (wrap at the end: harmless, never read)
            const int nk = (kt + 1) & (NT - 1);
            tdm_load_2d((unsigned)(unsigned long long)(void*)&Kt[cb ^ 1][0],
                        K + kg + (long)nk * 32 * DH, DH, LDIM, DH, 32, DH);
            tdm_load_2d((unsigned)(unsigned long long)(void*)&Vt[cb ^ 1][0],
                        VT + vg + (long)nk * 32, LDIM, DH, 32, DH, LDIM);
            __builtin_amdgcn_s_wait_tensorcnt(2);   // tile kt pair has landed
        }
        __syncthreads();
        const _Float16* Ktc = &Kt[cb][0];
        const _Float16* Vtc = &Vt[cb][0];

        // S^T tiles: t=0 -> keys 0..15, t=1 -> keys 16..31 (A operand = K rows)
        v8f st[2];
#pragma unroll
        for (int t = 0; t < 2; ++t) {
#pragma unroll
            for (int r = 0; r < 8; ++r) st[t][r] = 0.f;
#pragma unroll
            for (int f = 0; f < 2; ++f) {
                AFrag kf;
                const _Float16* p = &Ktc[(t * 16 + ln) * DH + f * 32 + kb0];
                kf.h[0] = *(const h8*)p;
                kf.h[1] = *(const h8*)(p + 16);
                st[t] = __builtin_amdgcn_wmma_f32_16x16x32_f16(
                    false, kf.v, false, qf[f].v, (short)0, st[t], false, false);
            }
        }

        // online softmax for query `ln` (base-2 logits, scale pre-folded into Q)
        float mx = st[0][0];
#pragma unroll
        for (int r = 1; r < 8; ++r) mx = fmaxf(mx, st[0][r]);
#pragma unroll
        for (int r = 0; r < 8; ++r) mx = fmaxf(mx, st[1][r]);
        mx = fmaxf(mx, __shfl_xor(mx, 16));    // combine key halves
        float mnew = fmaxf(m, mx);
        float corr = __builtin_amdgcn_exp2f(m - mnew);
        m = mnew;

        AFrag pf;                              // P B-fragment, built fully in-lane
        float rs = 0.f;
#pragma unroll
        for (int r = 0; r < 8; ++r) {
            float p0 = __builtin_amdgcn_exp2f(st[0][r] - m);
            float p1 = __builtin_amdgcn_exp2f(st[1][r] - m);
            rs += p0 + p1;
            pf.v[r]     = (_Float16)p0;        // keys kb0..kb0+7   (chunk 0)
            pf.v[8 + r] = (_Float16)p1;        // keys kb0+16..+23  (chunk 1)
        }
        rs += __shfl_xor(rs, 16);
        l = l * corr + rs;

#pragma unroll
        for (int n = 0; n < 4; ++n)
#pragma unroll
            for (int r = 0; r < 8; ++r) acc[n][r] *= corr;

        // out^T tiles: A operand = V^T rows (dh), B operand = P
#pragma unroll
        for (int n = 0; n < 4; ++n) {
            AFrag vf;
            const _Float16* p = &Vtc[(n * 16 + ln) * 32 + kb0];
            vf.h[0] = *(const h8*)p;
            vf.h[1] = *(const h8*)(p + 16);
            acc[n] = __builtin_amdgcn_wmma_f32_16x16x32_f16(
                false, vf.v, false, pf.v, (short)0, acc[n], false, false);
        }
        __syncthreads();   // all reads of buf cb done before tile kt+2 overwrites it
    }

    // epilogue: lane `ln` (both halves) owns query row qrow0+ln; regs are dh.
    const float inv = 1.0f / l;
    const int qrow = qrow0 + ln;
    float* op = out + ((long)((bh >> 3) * LDIM + qrow)) * DDIM + (bh & 7) * DH;
#pragma unroll
    for (int n = 0; n < 4; ++n) {
        float4 o0 = make_float4(acc[n][0] * inv, acc[n][1] * inv,
                                acc[n][2] * inv, acc[n][3] * inv);
        float4 o1 = make_float4(acc[n][4] * inv, acc[n][5] * inv,
                                acc[n][6] * inv, acc[n][7] * inv);
        *(float4*)(op + n * 16 + 8 * hiL)     = o0;
        *(float4*)(op + n * 16 + 8 * hiL + 4) = o1;
    }
}

extern "C" void kernel_launch(void* const* d_in, const int* in_sizes, int n_in,
                              void* d_out, int out_size, void* d_ws, size_t ws_size,
                              hipStream_t stream) {
    const float* x  = (const float*)d_in[0];
    const float* vf = (const float*)d_in[1];
    const float* wf = (const float*)d_in[2];
    float* out = (float*)d_out;

    const size_t nqkv = (size_t)BDIM * HDIM * LDIM * DH;
    _Float16* Qw  = (_Float16*)d_ws;
    _Float16* Kw  = Qw + nqkv;
    _Float16* VTw = Kw + nqkv;

    wavelet_qkv_kernel<<<BDIM * DDIM, 256, 0, stream>>>(x, vf, wf, Qw, Kw, VTw);
    flash_attn_kernel<<<BDIM * HDIM * (LDIM / 128), 256, 0, stream>>>(Qw, Kw, VTw, out);
}
